// QEffDeepseekV3Attention_63032940036293
// MI455X (gfx1250) — matrix operand order repaired and checked by reading, not compiled, not run on hardware
//
#include <hip/hip_runtime.h>
#include <hip/hip_bf16.h>
#include <cstdint>
#include <cstddef>

// ---------------- problem constants ----------------
#define S_LEN 2048
#define HIDN  2048
#define NH    16
#define DNOPE 128
#define DROT  64
#define DVAL  128
#define DQH   192   // DNOPE + DROT
#define QLORA 1536
#define KVLORA 512
#define SCALE_ATTN 0.07216878364870323f   // 192^-0.5

typedef __attribute__((ext_vector_type(16))) __bf16 v16bf;
typedef __attribute__((ext_vector_type(8)))  __bf16 v8bf;
typedef __attribute__((ext_vector_type(8)))  float  v8f;
typedef __attribute__((ext_vector_type(4)))  unsigned int u32x4;
typedef __attribute__((ext_vector_type(8)))  int i32x8;
typedef __attribute__((ext_vector_type(4)))  int i32x4;

#ifndef __has_builtin
#define __has_builtin(x) 0
#endif
#if __has_builtin(__builtin_amdgcn_tensor_load_to_lds) && \
    __has_builtin(__builtin_amdgcn_s_wait_tensorcnt)
#define HAVE_TDM 1
#else
#define HAVE_TDM 0
#endif

#define WMMA_BF16(a, b, c) \
  __builtin_amdgcn_wmma_f32_16x16x32_bf16(false, (a), false, (b), (short)0, (c), false, false)

__device__ __forceinline__ __bf16 f2bf(float f) {
  unsigned u = __builtin_bit_cast(unsigned, f);
  unsigned r = (u + 0x7FFFu + ((u >> 16) & 1u)) >> 16;
  unsigned short h = (unsigned short)r;
  return __builtin_bit_cast(__bf16, h);
}

__device__ __forceinline__ v16bf frag_cat(v8bf lo, v8bf hi) {
  v16bf r;
#pragma unroll
  for (int i = 0; i < 8; ++i) { r[i] = lo[i]; r[i + 8] = hi[i]; }
  return r;
}

// A-fragment (16x32 bf16, MxK): lane L -> row = L&15; lanes 0-15 hold K 0..7,16..23,
// lanes 16-31 hold K 8..15,24..31 (CDNA5 ISA 16-bit A layout).
__device__ __forceinline__ v16bf load_frag_a(const __bf16* t, int stride, int lane) {
  int row = lane & 15, half = lane >> 4;
  const __bf16* p = t + row * stride + half * 8;
  v8bf lo = *(const v8bf*)p;
  v8bf hi = *(const v8bf*)(p + 16);
  return frag_cat(lo, hi);
}

// B-fragment (32x16 bf16, KxN) from an LDS tile stored [N][K]: lane L -> col = L&15,
// lanes 0-15 hold K 0..7,16..23, lanes 16-31 hold K 8..15,24..31 (matches A).
__device__ __forceinline__ v16bf load_frag_b(const __bf16* t, int stride, int lane) {
  int col = lane & 15, half = lane >> 4;
  const __bf16* p = t + col * stride + half * 8;
  v8bf lo = *(const v8bf*)p;
  v8bf hi = *(const v8bf*)(p + 16);
  return frag_cat(lo, hi);
}

#if HAVE_TDM
// LDS generic address: low 32 bits are the LDS byte offset (aperture layout).
__device__ __forceinline__ unsigned lds_off(const void* p) {
  return (unsigned)(uintptr_t)p;
}

// Issue one 2-D TDM tile load: tile1 rows x tile0 bf16 elems, row stride stride0
// (elements). OOB rows/cols (beyond td0/td1) are zero-filled by the TDM.
// If pad: insert 4 dwords of LDS padding every 16 dwords (64B rows -> stride 40 bf16).
__device__ __forceinline__ void tdm_load_tile(unsigned ldsaddr, const void* gptr,
                                              unsigned td0, unsigned td1,
                                              unsigned tile0, unsigned tile1,
                                              unsigned long long stride0, bool pad) {
  unsigned long long ga = (unsigned long long)(uintptr_t)gptr;
  u32x4 g0;
  g0[0] = 1u;                                     // count=1, no gather
  g0[1] = ldsaddr;                                // lds_addr (bytes)
  g0[2] = (unsigned)(ga & 0xffffffffu);           // global_addr[31:0]
  g0[3] = (unsigned)((ga >> 32) & 0x1ffffffu) | (2u << 30);  // [56:32] | type=2
  unsigned w0 = (1u << 16);                       // data_size = 2 bytes
  if (pad) w0 |= (1u << 20) | (3u << 22) | (3u << 25);  // pad_en, every 16dw, +4dw
  i32x8 g1;
  g1[0] = (int)w0;
  g1[1] = (int)((td0 & 0xffffu) << 16);                       // tensor_dim0[15:0]
  g1[2] = (int)((td0 >> 16) | ((td1 & 0xffffu) << 16));       // td0[31:16]|td1[15:0]
  g1[3] = (int)((td1 >> 16) | ((tile0 & 0xffffu) << 16));     // td1[31:16]|tile0
  g1[4] = (int)(tile1 & 0xffffu);                             // tile1 (tile2=0)
  g1[5] = (int)(unsigned)(stride0 & 0xffffffffu);             // dim0 stride lo
  g1[6] = (int)(unsigned)((stride0 >> 32) & 0xffffu);         // dim0 stride hi
  g1[7] = 0;                                                  // dim1 stride (2D)
  i32x4 z4; z4[0] = 0; z4[1] = 0; z4[2] = 0; z4[3] = 0;
  i32x8 z8;
#pragma unroll
  for (int i = 0; i < 8; ++i) z8[i] = 0;
  __builtin_amdgcn_tensor_load_to_lds(g0, g1, z4, z4, z8, 0);
}
#endif

// ---------------- cast fp32 -> bf16 ----------------
__global__ void cast_f32_bf16(const float* __restrict__ x, __bf16* __restrict__ y, size_t n) {
  size_t i = (size_t)blockIdx.x * blockDim.x + threadIdx.x;
  size_t stride = (size_t)gridDim.x * blockDim.x;
  for (; i < n; i += stride) y[i] = f2bf(x[i]);
}

// ---------------- RMSNorm (row per block) ----------------
__global__ void __launch_bounds__(256)
rmsnorm_bf16(const float* __restrict__ x, const float* __restrict__ w,
             __bf16* __restrict__ y, int D, int xstride) {
  int row = blockIdx.x;
  const float* xr = x + (size_t)row * xstride;
  float ss = 0.f;
  for (int i = threadIdx.x; i < D; i += 256) { float v = xr[i]; ss += v * v; }
#pragma unroll
  for (int m = 16; m >= 1; m >>= 1) ss += __shfl_xor(ss, m, 32);
  __shared__ float red[8];
  if ((threadIdx.x & 31) == 0) red[threadIdx.x >> 5] = ss;
  __syncthreads();
  float tot = 0.f;
#pragma unroll
  for (int i = 0; i < 8; ++i) tot += red[i];
  float r = rsqrtf(tot / (float)D + 1e-6f);
  for (int i = threadIdx.x; i < D; i += 256) y[(size_t)row * D + i] = f2bf(xr[i] * r * w[i]);
}

// ---------------- WMMA GEMM: C[M,N] = A[M,K] (bf16) x W[N,K]^T (bf16), fp32 out ----
// TDM double-buffered: wave 0 issues next tile's tensor_load_to_lds while all
// 8 waves run WMMA on the current buffer.
template <bool NCHECK>
__global__ void __launch_bounds__(256)
gemm_bf16_kernel(const __bf16* __restrict__ A, const __bf16* __restrict__ W,
                 float* __restrict__ C, int M, int N, int K) {
  constexpr int STR = 40;            // 32 bf16 + 8 pad (TDM pad: 16dw + 4dw)
  __shared__ __bf16 As[2][128 * STR];
  __shared__ __bf16 Ws[2][128 * STR];
  int tid  = threadIdx.x;
  int lane = tid & 31;
  int wave = tid >> 5;
  int bm = blockIdx.y * 128;
  int bn = blockIdx.x * 128;
  int wr = (wave >> 1) * 32;         // 4x2 wave grid: 32 rows x 64 cols per wave
  int wc = (wave & 1) * 64;

  v8f acc[2][4];
#pragma unroll
  for (int i = 0; i < 2; ++i)
#pragma unroll
    for (int j = 0; j < 4; ++j)
#pragma unroll
      for (int v = 0; v < 8; ++v) acc[i][j][v] = 0.f;

  int nk = K >> 5;
#if HAVE_TDM
  if (tid < 32) {
    tdm_load_tile(lds_off(&As[0][0]), A + (size_t)bm * K, (unsigned)K, (unsigned)(M - bm),
                  32u, 128u, (unsigned long long)K, true);
    tdm_load_tile(lds_off(&Ws[0][0]), W + (size_t)bn * K, (unsigned)K, (unsigned)(N - bn),
                  32u, 128u, (unsigned long long)K, true);
  }
#endif
  for (int t = 0; t < nk; ++t) {
    int cur = t & 1;
    int kk = t << 5;
#if HAVE_TDM
    (void)kk;
    if (tid < 32) __builtin_amdgcn_s_wait_tensorcnt(0);
    __syncthreads();                            // current tile visible to all waves
    if (tid < 32 && t + 1 < nk) {
      int kn = kk + 32;
      tdm_load_tile(lds_off(&As[cur ^ 1][0]), A + (size_t)bm * K + kn,
                    (unsigned)(K - kn), (unsigned)(M - bm), 32u, 128u,
                    (unsigned long long)K, true);
      tdm_load_tile(lds_off(&Ws[cur ^ 1][0]), W + (size_t)bn * K + kn,
                    (unsigned)(K - kn), (unsigned)(N - bn), 32u, 128u,
                    (unsigned long long)K, true);
    }
#else
#pragma unroll
    for (int c = tid; c < 512; c += 256) {      // 128 rows x 4 chunks of 8 bf16
      int r = c >> 2, q = c & 3;
      *(v8bf*)&As[cur][r * STR + q * 8] =
          *(const v8bf*)(A + (size_t)(bm + r) * K + kk + q * 8);
      v8bf wv;
      int gn = bn + r;
      if (!NCHECK || gn < N) {
        wv = *(const v8bf*)(W + (size_t)gn * K + kk + q * 8);
      } else {
#pragma unroll
        for (int e = 0; e < 8; ++e) wv[e] = f2bf(0.f);
      }
      *(v8bf*)&Ws[cur][r * STR + q * 8] = wv;
    }
    __syncthreads();
#endif
    v16bf bfr[4];
#pragma unroll
    for (int j = 0; j < 4; ++j)
      bfr[j] = load_frag_b(&Ws[cur][(wc + j * 16) * STR], STR, lane);
#pragma unroll
    for (int i = 0; i < 2; ++i) {
      v16bf af = load_frag_a(&As[cur][(wr + i * 16) * STR], STR, lane);
#pragma unroll
      for (int j = 0; j < 4; ++j) acc[i][j] = WMMA_BF16(af, bfr[j], acc[i][j]);
    }
    __syncthreads();                            // done reading before refill
  }

  int cl = lane & 15, hf = lane >> 4;
#pragma unroll
  for (int i = 0; i < 2; ++i)
#pragma unroll
    for (int j = 0; j < 4; ++j) {
      int col = bn + wc + j * 16 + cl;
      if (NCHECK && col >= N) continue;
#pragma unroll
      for (int v = 0; v < 8; ++v) {
        int row = bm + wr + i * 16 + v + 8 * hf;
        C[(size_t)row * N + col] = acc[i][j][v];
      }
    }
}

// ---------------- RoPE / pack kernels ----------------
__global__ void pack_q_kernel(const float* __restrict__ q, const float* __restrict__ cosb,
                              const float* __restrict__ sinb, const int* __restrict__ pos,
                              __bf16* __restrict__ Qb) {
  int s = blockIdx.x, h = blockIdx.y, d = threadIdx.x;
  const float* qr = q + ((size_t)s * NH + h) * DQH;
  float v;
  if (d < DNOPE) {
    v = qr[d];
  } else {
    int j = d - DNOPE;
    int p = pos[s];
    const float* cr = cosb + (size_t)p * DROT;
    const float* sr = sinb + (size_t)p * DROT;
    const int half = DROT / 2;
    if (j < half) v = qr[DNOPE + 2 * j] * cr[j] - qr[DNOPE + 2 * j + 1] * sr[j];
    else { int jj = j - half; v = qr[DNOPE + 2 * jj + 1] * cr[j] + qr[DNOPE + 2 * jj] * sr[j]; }
  }
  Qb[((size_t)h * S_LEN + s) * DQH + d] = f2bf(v);
}

__global__ void pack_k_kernel(const float* __restrict__ kv, const float* __restrict__ ckv,
                              const float* __restrict__ cosb, const float* __restrict__ sinb,
                              const int* __restrict__ pos, __bf16* __restrict__ Kb) {
  int s = blockIdx.x, h = blockIdx.y, d = threadIdx.x;
  float v;
  if (d < DNOPE) {
    v = kv[((size_t)s * NH + h) * (DNOPE + DVAL) + d];
  } else {
    int j = d - DNOPE;
    const float* kr = ckv + (size_t)s * (KVLORA + DROT) + KVLORA;
    int p = pos[s];
    const float* cr = cosb + (size_t)p * DROT;
    const float* sr = sinb + (size_t)p * DROT;
    const int half = DROT / 2;
    if (j < half) v = kr[2 * j] * cr[j] - kr[2 * j + 1] * sr[j];
    else { int jj = j - half; v = kr[2 * jj + 1] * cr[j] + kr[2 * jj] * sr[j]; }
  }
  Kb[((size_t)h * S_LEN + s) * DQH + d] = f2bf(v);
}

__global__ void pack_v_kernel(const float* __restrict__ kv, __bf16* __restrict__ Vt) {
  int s = blockIdx.x, h = blockIdx.y, d = threadIdx.x;
  float v = kv[((size_t)s * NH + h) * (DNOPE + DVAL) + DNOPE + d];
  Vt[((size_t)h * DVAL + d) * S_LEN + s] = f2bf(v);   // V transposed: [h][d][s]
}

// ---------------- Flash-style causal attention (WMMA + TDM tile fills) -------
// block = 128 threads (4 waves): one head, 64 query rows (16 per wave).
// Key tiles of 32; per tile: 12 WMMA for QK^T (192 = 6 ksteps x 2 ntiles) + 8 for PV.
__global__ void __launch_bounds__(128)
attn_kernel(const __bf16* __restrict__ Qb, const __bf16* __restrict__ Kb,
            const __bf16* __restrict__ Vt, __bf16* __restrict__ attn) {
  __shared__ __bf16 Qs[64 * DQH];    // 64 x 192, unpadded (TDM pad needs pow2 interval)
  __shared__ __bf16 Ks[32 * DQH];    // 32 x 192
  __shared__ __bf16 Vts[128 * 40];   // [d=128][key=32] + 8 pad (TDM 16dw+4dw)
  __shared__ __bf16 Ps[4 * 16 * 40]; // per-wave 16x32 P staging (+8 pad)
  int h = blockIdx.y;
  int qt = blockIdx.x;
  int qbase = qt * 64;
  int tid = threadIdx.x, lane = tid & 31, wid = tid >> 5;
  int cl = lane & 15, hf = lane >> 4;

  const __bf16* Qg = Qb + ((size_t)h * S_LEN + qbase) * DQH;
#if HAVE_TDM
  if (tid < 32) {
    tdm_load_tile(lds_off(&Qs[0]), Qg, (unsigned)DQH, 64u, (unsigned)DQH, 64u,
                  (unsigned long long)DQH, false);
  }
#else
  for (int c = tid; c < 64 * 24; c += 128) {       // 192/8 = 24 chunks/row
    int r = c / 24, q = c % 24;
    *(v8bf*)&Qs[r * DQH + q * 8] = *(const v8bf*)(Qg + (size_t)r * DQH + q * 8);
  }
#endif

  float m[8], l[8];
  v8f o[8];
#pragma unroll
  for (int v = 0; v < 8; ++v) { m[v] = -1e30f; l[v] = 0.f; }
#pragma unroll
  for (int j = 0; j < 8; ++j)
#pragma unroll
    for (int v = 0; v < 8; ++v) o[j][v] = 0.f;

  int nkt = 2 * qt + 2;                            // causal: keys 0 .. qbase+63
  for (int kt = 0; kt < nkt; ++kt) {
    int kbase = kt * 32;
    __syncthreads();                               // previous tile fully consumed
    const __bf16* Kg = Kb + ((size_t)h * S_LEN + kbase) * DQH;
    const __bf16* Vg = Vt + (size_t)h * DVAL * S_LEN + kbase;
#if HAVE_TDM
    if (tid < 32) {
      tdm_load_tile(lds_off(&Ks[0]), Kg, (unsigned)DQH, 32u, (unsigned)DQH, 32u,
                    (unsigned long long)DQH, false);
      tdm_load_tile(lds_off(&Vts[0]), Vg, (unsigned)(S_LEN - kbase), 128u, 32u, 128u,
                    (unsigned long long)S_LEN, true);
      __builtin_amdgcn_s_wait_tensorcnt(0);
    }
#else
    for (int c = tid; c < 32 * 24; c += 128) {
      int r = c / 24, q = c % 24;
      *(v8bf*)&Ks[r * DQH + q * 8] = *(const v8bf*)(Kg + (size_t)r * DQH + q * 8);
    }
    for (int c = tid; c < 128 * 4; c += 128) {
      int r = c >> 2, q = c & 3;
      *(v8bf*)&Vts[r * 40 + q * 8] = *(const v8bf*)(Vg + (size_t)r * S_LEN + q * 8);
    }
#endif
    __syncthreads();

    // S = Q K^T for this wave's 16 rows x 32 keys
    v8f s0, s1;
#pragma unroll
    for (int v = 0; v < 8; ++v) { s0[v] = 0.f; s1[v] = 0.f; }
#pragma unroll
    for (int kk = 0; kk < DQH; kk += 32) {
      v16bf a  = load_frag_a(&Qs[(wid * 16) * DQH + kk], DQH, lane);
      v16bf b0 = load_frag_b(&Ks[kk], DQH, lane);
      v16bf b1 = load_frag_b(&Ks[16 * DQH + kk], DQH, lane);
      s0 = WMMA_BF16(a, b0, s0);
      s1 = WMMA_BF16(a, b1, s1);
    }

    bool domask = (kbase + 31) > (qbase + wid * 16);
#pragma unroll
    for (int v = 0; v < 8; ++v) {
      int rowg = qbase + wid * 16 + v + 8 * hf;
      float x0 = s0[v] * SCALE_ATTN;
      float x1 = s1[v] * SCALE_ATTN;
      if (domask) {
        if (kbase + cl > rowg)      x0 = -10000.f;
        if (kbase + 16 + cl > rowg) x1 = -10000.f;
      }
      float mx = fmaxf(x0, x1);
#pragma unroll
      for (int sft = 8; sft >= 1; sft >>= 1) mx = fmaxf(mx, __shfl_xor(mx, sft, 32));
      float mnew = fmaxf(m[v], mx);
      float p0 = __expf(x0 - mnew);
      float p1 = __expf(x1 - mnew);
      float rs = p0 + p1;
#pragma unroll
      for (int sft = 8; sft >= 1; sft >>= 1) rs += __shfl_xor(rs, sft, 32);
      float corr = __expf(m[v] - mnew);
      l[v] = l[v] * corr + rs;
      m[v] = mnew;
#pragma unroll
      for (int j = 0; j < 8; ++j) o[j][v] *= corr;
      int pr = v + 8 * hf;                          // C-layout row -> P staging row
      Ps[(wid * 16 + pr) * 40 + cl]      = f2bf(p0);
      Ps[(wid * 16 + pr) * 40 + 16 + cl] = f2bf(p1);
    }

    // O += P V  (contraction over 32 keys = one WMMA kstep; Vts is [d][key])
    v16bf pa = load_frag_a(&Ps[(wid * 16) * 40], 40, lane);
#pragma unroll
    for (int j = 0; j < 8; ++j) {
      v16bf vb = load_frag_b(&Vts[(j * 16) * 40], 40, lane);
      o[j] = WMMA_BF16(pa, vb, o[j]);
    }
  }

#pragma unroll
  for (int j = 0; j < 8; ++j)
#pragma unroll
    for (int v = 0; v < 8; ++v) {
      int row = qbase + wid * 16 + v + 8 * hf;
      int col = h * DVAL + j * 16 + cl;
      attn[(size_t)row * (NH * DVAL) + col] = f2bf(o[j][v] / l[v]);
    }
}

// ---------------- launcher ----------------
extern "C" void kernel_launch(void* const* d_in, const int* in_sizes, int n_in,
                              void* d_out, int out_size, void* d_ws, size_t ws_size,
                              hipStream_t stream) {
  (void)in_sizes; (void)n_in; (void)out_size; (void)ws_size;
  const float* hidden  = (const float*)d_in[0];
  const float* q_a_w   = (const float*)d_in[1];
  const float* q_a_ln  = (const float*)d_in[2];
  const float* q_b_w   = (const float*)d_in[3];
  const float* kv_a_w  = (const float*)d_in[4];
  const float* kv_a_ln = (const float*)d_in[5];
  const float* kv_b_w  = (const float*)d_in[6];
  const float* o_w     = (const float*)d_in[7];
  const float* cosb    = (const float*)d_in[8];
  const float* sinb    = (const float*)d_in[9];
  const int*   pos     = (const int*)d_in[11];
  float* out = (float*)d_out;

  char* ws = (char*)d_ws;
  size_t off = 0;
  auto alloc = [&](size_t bytes) -> void* {
    off = (off + 255) & ~(size_t)255;
    void* p = ws + off;
    off += bytes;
    return p;
  };

  __bf16* hid_bf  = (__bf16*)alloc((size_t)S_LEN * HIDN * 2);
  __bf16* wqa_bf  = (__bf16*)alloc((size_t)QLORA * HIDN * 2);
  __bf16* wqb_bf  = (__bf16*)alloc((size_t)NH * DQH * QLORA * 2);
  __bf16* wkva_bf = (__bf16*)alloc((size_t)(KVLORA + DROT) * HIDN * 2);
  __bf16* wkvb_bf = (__bf16*)alloc((size_t)NH * (DNOPE + DVAL) * KVLORA * 2);
  __bf16* wo_bf   = (__bf16*)alloc((size_t)HIDN * NH * DVAL * 2);
  float*  qa_f    = (float*)alloc((size_t)S_LEN * QLORA * 4);
  __bf16* qa_bf   = (__bf16*)alloc((size_t)S_LEN * QLORA * 2);
  float*  q_f     = (float*)alloc((size_t)S_LEN * NH * DQH * 4);
  float*  ckv_f   = (float*)alloc((size_t)S_LEN * (KVLORA + DROT) * 4);
  __bf16* ckvn_bf = (__bf16*)alloc((size_t)S_LEN * KVLORA * 2);
  float*  kv_f    = (float*)alloc((size_t)S_LEN * NH * (DNOPE + DVAL) * 4);
  __bf16* Qb      = (__bf16*)alloc((size_t)NH * S_LEN * DQH * 2);
  __bf16* Kb      = (__bf16*)alloc((size_t)NH * S_LEN * DQH * 2);
  __bf16* Vt      = (__bf16*)alloc((size_t)NH * DVAL * S_LEN * 2);
  __bf16* attn_bf = (__bf16*)alloc((size_t)S_LEN * NH * DVAL * 2);

  auto cast = [&](const float* x, __bf16* y, size_t n) {
    int blocks = (int)((n + 256 * 8 - 1) / (256 * 8));
    cast_f32_bf16<<<blocks, 256, 0, stream>>>(x, y, n);
  };
  auto gemm = [&](const __bf16* A, const __bf16* W, float* C, int M, int N, int K,
                  bool ncheck) {
    dim3 g((N + 127) / 128, (M + 127) / 128);
    if (ncheck) gemm_bf16_kernel<true><<<g, 256, 0, stream>>>(A, W, C, M, N, K);
    else        gemm_bf16_kernel<false><<<g, 256, 0, stream>>>(A, W, C, M, N, K);
  };

  cast(hidden, hid_bf,  (size_t)S_LEN * HIDN);
  cast(q_a_w,  wqa_bf,  (size_t)QLORA * HIDN);
  cast(q_b_w,  wqb_bf,  (size_t)NH * DQH * QLORA);
  cast(kv_a_w, wkva_bf, (size_t)(KVLORA + DROT) * HIDN);
  cast(kv_b_w, wkvb_bf, (size_t)NH * (DNOPE + DVAL) * KVLORA);
  cast(o_w,    wo_bf,   (size_t)HIDN * NH * DVAL);

  // q = rmsnorm(hidden @ q_a_w^T) @ q_b_w^T
  gemm(hid_bf, wqa_bf, qa_f, S_LEN, QLORA, HIDN, false);
  rmsnorm_bf16<<<S_LEN, 256, 0, stream>>>(qa_f, q_a_ln, qa_bf, QLORA, QLORA);
  gemm(qa_bf, wqb_bf, q_f, S_LEN, NH * DQH, QLORA, false);

  // ckv = hidden @ kv_a_w^T ; kv = rmsnorm(ckv[:,:512]) @ kv_b_w^T
  gemm(hid_bf, wkva_bf, ckv_f, S_LEN, KVLORA + DROT, HIDN, true);
  rmsnorm_bf16<<<S_LEN, 256, 0, stream>>>(ckv_f, kv_a_ln, ckvn_bf, KVLORA, KVLORA + DROT);
  gemm(ckvn_bf, wkvb_bf, kv_f, S_LEN, NH * (DNOPE + DVAL), KVLORA, false);

  // RoPE + attention-friendly packing
  pack_q_kernel<<<dim3(S_LEN, NH), DQH, 0, stream>>>(q_f, cosb, sinb, pos, Qb);
  pack_k_kernel<<<dim3(S_LEN, NH), DQH, 0, stream>>>(kv_f, ckv_f, cosb, sinb, pos, Kb);
  pack_v_kernel<<<dim3(S_LEN, NH), DVAL, 0, stream>>>(kv_f, Vt);

  // causal flash attention
  attn_kernel<<<dim3(S_LEN / 64, NH), 128, 0, stream>>>(Qb, Kb, Vt, attn_bf);

  // output projection
  gemm(attn_bf, wo_bf, out, S_LEN, HIDN, NH * DVAL, false);
}
